// GATModel_77945066488470
// MI455X (gfx1250) — matrix operand order repaired and checked by reading, not compiled, not run on hardware
//
#include <hip/hip_runtime.h>
#include <hip/hip_bf16.h>

// ---------------------------------------------------------------------------
// Fused GAT model for MI455X (gfx1250, wave32, WMMA).
// One workgroup (256 threads = 8 wave32) per graph (G=2048, N=48, C=128).
// Each GAT layer: LDS-staged f16 WMMA GEMM -> per-graph CSR edge softmax ->
// aggregation -> fused per-graph LayerNorm (+ReLU) -> single global write.
// Final kernel: WMMA GEMM (fc1) + fc2 + node softmax + attention avg + top-4.
// LDS arrays use padded strides to avoid 16/32-way bank conflicts
// (f16 row stride KIN+2 -> odd dword stride; f32 row stride 132).
// ---------------------------------------------------------------------------

#define G_GRAPHS 2048
#define N_NODES  48
#define HID      128
#define DEG      8
#define EPG      (N_NODES * DEG)   // 384 edges per graph
#define OUT_PER_G 520              // 4*128 pooled + 8 att
#define HSLD     132               // padded f32 row stride
#define ULD      9                 // padded logits stride

typedef _Float16 f16;
typedef __attribute__((ext_vector_type(16))) _Float16 v16h;
typedef __attribute__((ext_vector_type(8)))  float    v8f;

__device__ __forceinline__ float leaky(float x) { return x > 0.f ? x : 0.2f * x; }

// A-matrix fragment: 16x32 f16 tile, rows striped in-lane; contiguous K order (same as B, not the ISA table order).
// lane m = lane&15, hi = lane>>4 ; dword p holds K = 2p + 16*hi
__device__ __forceinline__ v16h load_a_frag(const f16* Xh, int ld, int m, int k0, int hi) {
  union { v16h v; unsigned u[8]; } A;
  const f16* ap = Xh + m * ld + k0 + 16 * hi;
#pragma unroll
  for (int p = 0; p < 8; ++p) {
    A.u[p] = *(const unsigned*)(ap + 2 * p);
  }
  return A.v;
}

// B-matrix fragment: 32x16 f16, column n per lane; W staged column-major in LDS
// so lane's K run (k0 + 16*hi .. +15) is contiguous. dword p holds K = 2p.
__device__ __forceinline__ v16h load_b_frag(const f16* Wt, int ld, int n, int k0, int hi) {
  union { v16h v; unsigned u[8]; } B;
  const f16* bp = Wt + n * ld + k0 + 16 * hi;
#pragma unroll
  for (int p = 0; p < 8; ++p) B.u[p] = *(const unsigned*)(bp + 2 * p);
  return B.v;
}

// ---------------------------------------------------------------------------
// Fused GAT layer. KIN = input feature count (64 or 128).
// ---------------------------------------------------------------------------
template <int KIN, bool RELU>
__global__ __launch_bounds__(256) void gat_layer(
    const float* __restrict__ hin, const int* __restrict__ g_esrc,
    const int* __restrict__ g_edst, const float* __restrict__ W,
    const float* __restrict__ a_s, const float* __restrict__ a_d,
    const float* __restrict__ bias, const float* __restrict__ lnw,
    const float* __restrict__ lnb, float* __restrict__ hout) {
  constexpr int LD = KIN + 2;          // padded f16 stride (odd dword stride)
  extern __shared__ char smem[];
  f16*   Wt   = (f16*)smem;            // [128][LD] col-major staged weights
  f16*   Xh   = Wt + 128 * LD;         // [48][LD]  f16 inputs
  float* Hs   = (float*)(Xh + 48 * LD);   // [48][HSLD] h = x @ W (f32)
  float* Ssrc = Hs + 48 * HSLD;        // [48] a_src . h
  float* Sdst = Ssrc + 48;             // [48] a_dst . h
  int*   Esrc = (int*)(Sdst + 48);     // [384]
  int*   Edst = Esrc + EPG;            // [384]
  int*   Cnt  = Edst + EPG;            // [48] in-degree
  int*   Off  = Cnt + 48;              // [49] CSR offsets
  int*   Woff = Off + 49;              // [48] scatter cursors
  int*   Csr  = Woff + 48;             // [384] incoming-src per dst
  float* Red  = (float*)(Csr + EPG);   // [16] LN partials

  const int g = blockIdx.x;
  const int tid = threadIdx.x;
  const int wv = tid >> 5, lane = tid & 31;
  const int base = g * N_NODES;

  // per-lane channel constants (c = lane + 32q)
  float asq[4], adq[4], bq[4], lwq[4], lbq[4];
#pragma unroll
  for (int q = 0; q < 4; ++q) {
    int c = lane + 32 * q;
    asq[q] = a_s[c]; adq[q] = a_d[c];
    bq[q] = bias[c]; lwq[q] = lnw[c]; lbq[q] = lnb[c];
  }

  // ---- phase 1: stage weights (transposed, f16), inputs (f16), edges ----
  for (int idx = tid; idx < KIN * 128; idx += 256) {
    int k = idx >> 7, n = idx & 127;
    Wt[n * LD + k] = (f16)W[idx];
  }
  for (int idx = tid; idx < 48 * KIN; idx += 256) {
    int r = idx / KIN, c = idx - r * KIN;
    Xh[r * LD + c] = (f16)hin[(size_t)base * KIN + idx];
  }
  for (int e = tid; e < EPG; e += 256) {
    Esrc[e] = g_esrc[g * EPG + e] - base;
    Edst[e] = g_edst[g * EPG + e] - base;
  }
  if (tid < 48) Cnt[tid] = 0;
  __syncthreads();

  // ---- phase 2: in-degree count + WMMA GEMM h = x @ W ----
  for (int e = tid; e < EPG; e += 256) atomicAdd(&Cnt[Edst[e]], 1);
  {
    const int n0 = wv * 16, hi = lane >> 4, nn = lane & 15;
    v16h bfr[KIN / 32];                 // hoist B fragments across M-tiles
#pragma unroll
    for (int ks = 0; ks < KIN / 32; ++ks)
      bfr[ks] = load_b_frag(Wt, LD, n0 + nn, ks * 32, hi);
#pragma unroll
    for (int mt = 0; mt < 3; ++mt) {
      const int m0 = mt * 16;
      v8f acc = {0.f, 0.f, 0.f, 0.f, 0.f, 0.f, 0.f, 0.f};
#pragma unroll
      for (int ks = 0; ks < KIN / 32; ++ks) {
        v16h a = load_a_frag(Xh, LD, m0 + nn, ks * 32, hi);
        acc = __builtin_amdgcn_wmma_f32_16x16x32_f16(false, a, false, bfr[ks],
                                                     (short)0, acc, false, false);
      }
#pragma unroll
      for (int v = 0; v < 8; ++v)
        Hs[(m0 + v + 8 * hi) * HSLD + n0 + nn] = acc[v];
    }
  }
  __syncthreads();

  // ---- phase 3: CSR prefix ----
  if (tid == 0) {
    int a = 0;
    for (int n = 0; n < 48; ++n) { Off[n] = a; Woff[n] = a; a += Cnt[n]; }
    Off[48] = a;
  }
  __syncthreads();

  // ---- phase 4: CSR scatter + attention logit pre-dots (lane-parallel) ----
  for (int e = tid; e < EPG; e += 256) {
    int p = atomicAdd(&Woff[Edst[e]], 1);
    Csr[p] = Esrc[e];
  }
#pragma unroll
  for (int t = 0; t < 6; ++t) {
    const int i = wv + 8 * t;
    float ss = 0.f, sd = 0.f;
#pragma unroll
    for (int q = 0; q < 4; ++q) {
      float v = Hs[i * HSLD + lane + 32 * q];
      ss += v * asq[q]; sd += v * adq[q];
    }
#pragma unroll
    for (int o = 16; o > 0; o >>= 1) { ss += __shfl_xor(ss, o, 32); sd += __shfl_xor(sd, o, 32); }
    if (lane == 0) { Ssrc[i] = ss; Sdst[i] = sd; }
  }
  __syncthreads();

  // ---- phase 5: per-node softmax-attention aggregation (wave wv: nodes wv+8t) ----
  float vreg[6][4];
#pragma unroll
  for (int t = 0; t < 6; ++t) {
    const int i = wv + 8 * t;
    const int e0 = Off[i], nE = Cnt[i];
    const float sdi = Sdst[i];
    const float es = leaky(Ssrc[i] + sdi);  // self-loop logit
    float mx = es;
    for (int j = lane; j < nE; j += 32)
      mx = fmaxf(mx, leaky(Ssrc[Csr[e0 + j]] + sdi));
#pragma unroll
    for (int o = 16; o > 0; o >>= 1) mx = fmaxf(mx, __shfl_xor(mx, o, 32));
    float z = (lane == 0) ? __expf(es - mx) : 0.f;
    for (int j = lane; j < nE; j += 32)
      z += __expf(leaky(Ssrc[Csr[e0 + j]] + sdi) - mx);
#pragma unroll
    for (int o = 16; o > 0; o >>= 1) z += __shfl_xor(z, o, 32);
    z += 1e-16f;
    const float iz = 1.f / z;
    const float ps = __expf(es - mx);
    float a0 = ps * Hs[i * HSLD + lane];
    float a1 = ps * Hs[i * HSLD + lane + 32];
    float a2 = ps * Hs[i * HSLD + lane + 64];
    float a3 = ps * Hs[i * HSLD + lane + 96];
    for (int j = 0; j < nE; ++j) {
      const int s = Csr[e0 + j];
      const float p = __expf(leaky(Ssrc[s] + sdi) - mx);
      const float* hp = Hs + s * HSLD + lane;
      a0 += p * hp[0];  a1 += p * hp[32];
      a2 += p * hp[64]; a3 += p * hp[96];
    }
    vreg[t][0] = a0 * iz + bq[0];
    vreg[t][1] = a1 * iz + bq[1];
    vreg[t][2] = a2 * iz + bq[2];
    vreg[t][3] = a3 * iz + bq[3];
  }

  // ---- phase 6: fused per-graph LayerNorm (+ReLU) on register values ----
  float S = 0.f, SS = 0.f;
#pragma unroll
  for (int t = 0; t < 6; ++t)
#pragma unroll
    for (int q = 0; q < 4; ++q) { float v = vreg[t][q]; S += v; SS += v * v; }
#pragma unroll
  for (int o = 16; o > 0; o >>= 1) { S += __shfl_xor(S, o, 32); SS += __shfl_xor(SS, o, 32); }
  if (lane == 0) { Red[wv] = S; Red[8 + wv] = SS; }
  __syncthreads();
  float St = 0.f, SSt = 0.f;
#pragma unroll
  for (int k = 0; k < 8; ++k) { St += Red[k]; SSt += Red[8 + k]; }
  const float mu  = St * (1.f / 6144.f);
  const float var = SSt * (1.f / 6144.f) - mu * mu;
  const float rst = rsqrtf(var + 1e-5f);
#pragma unroll
  for (int t = 0; t < 6; ++t) {
    const int i = wv + 8 * t;
    float* op = hout + (size_t)(base + i) * 128;
#pragma unroll
    for (int q = 0; q < 4; ++q) {
      float v = (vreg[t][q] - mu) * rst * lwq[q] + lbq[q];
      if (RELU) v = fmaxf(v, 0.f);
      op[lane + 32 * q] = v;
    }
  }
}

// ---------------------------------------------------------------------------
// Final: TopK pooling + attention pooling, fused per graph.
// out[g] = [ h[top4]*score (512) | att-pooled avg (8) ]
// ---------------------------------------------------------------------------
__global__ __launch_bounds__(256) void final_pool(
    const float* __restrict__ h3, const float* __restrict__ pw,
    const float* __restrict__ f1w, const float* __restrict__ f1b,
    const float* __restrict__ f2w, const float* __restrict__ f2b,
    float* __restrict__ out) {
  constexpr int LD = 130;               // padded f16 stride
  extern __shared__ char smem[];
  f16*   Wt = (f16*)smem;               // [128][LD] fc1 col-major f16
  f16*   Xh = Wt + 128 * LD;            // [48][LD]  h3 f16
  float* Hs = (float*)(Xh + 48 * LD);   // [48][HSLD] h3 f32
  float* Ts = Hs + 48 * HSLD;           // [48][HSLD] t = tanh(h@fc1+b)
  float* U  = Ts + 48 * HSLD;           // [48][ULD]  logits
  float* Rs = U + 48 * ULD;             // [48] row sums of h3
  float* Sc = Rs + 48;                  // [48] topk scores
  float* Tv = Sc + 48;                  // [4]
  int*   Ti = (int*)(Tv + 4);           // [4]

  const int g = blockIdx.x;
  const int tid = threadIdx.x;
  const int wv = tid >> 5, lane = tid & 31;
  const int base = g * N_NODES;

  for (int idx = tid; idx < 128 * 128; idx += 256) {
    int k = idx >> 7, n = idx & 127;
    Wt[n * LD + k] = (f16)f1w[idx];
  }
  for (int idx = tid; idx < 48 * 128; idx += 256) {
    int r = idx >> 7, c = idx & 127;
    float v = h3[(size_t)base * 128 + idx];
    Hs[r * HSLD + c] = v; Xh[r * LD + c] = (f16)v;
  }
  __syncthreads();

  // t = tanh(h3 @ fc1 + b1) via WMMA
  {
    const int n0 = wv * 16, hi = lane >> 4, nn = lane & 15;
    v16h bfr[4];
#pragma unroll
    for (int ks = 0; ks < 4; ++ks)
      bfr[ks] = load_b_frag(Wt, LD, n0 + nn, ks * 32, hi);
#pragma unroll
    for (int mt = 0; mt < 3; ++mt) {
      const int m0 = mt * 16;
      v8f acc = {0.f, 0.f, 0.f, 0.f, 0.f, 0.f, 0.f, 0.f};
#pragma unroll
      for (int ks = 0; ks < 4; ++ks) {
        v16h a = load_a_frag(Xh, LD, m0 + nn, ks * 32, hi);
        acc = __builtin_amdgcn_wmma_f32_16x16x32_f16(false, a, false, bfr[ks],
                                                     (short)0, acc, false, false);
      }
#pragma unroll
      for (int v = 0; v < 8; ++v) {
        int m = m0 + v + 8 * hi, n = n0 + nn;
        Ts[m * HSLD + n] = tanhf(acc[v] + f1b[n]);
      }
    }
  }
  __syncthreads();

  // u = t @ fc2 + b2 (lane-parallel per node, wave wv: nodes wv+8t)
#pragma unroll
  for (int t = 0; t < 6; ++t) {
    const int i = wv + 8 * t;
    float ua[8] = {0.f, 0.f, 0.f, 0.f, 0.f, 0.f, 0.f, 0.f};
#pragma unroll
    for (int q = 0; q < 4; ++q) {
      const int c = lane + 32 * q;
      const float tv = Ts[i * HSLD + c];
#pragma unroll
      for (int a = 0; a < 8; ++a) ua[a] += tv * f2w[c * 8 + a];
    }
#pragma unroll
    for (int a = 0; a < 8; ++a) {
      float s = ua[a];
#pragma unroll
      for (int o = 16; o > 0; o >>= 1) s += __shfl_xor(s, o, 32);
      if (lane == 0) U[i * ULD + a] = s + f2b[a];
    }
  }
  // row sums + topk scores (HSLD padding keeps this ~2-way at worst)
  if (tid < 48) {
    float r = 0.f, sc = 0.f, n2 = 0.f;
#pragma unroll 4
    for (int c = 0; c < 128; ++c) {
      float v = Hs[tid * HSLD + c];
      r += v; sc += v * pw[c]; n2 += pw[c] * pw[c];
    }
    Rs[tid] = r;
    Sc[tid] = tanhf(sc * rsqrtf(n2));
  }
  __syncthreads();

  // per-head softmax over nodes fused with avg = (att . rowsum)/8 (wave wv == head)
  {
    const int a = wv;
    float v0 = U[lane * ULD + a];
    float v1 = (lane < 16) ? U[(lane + 32) * ULD + a] : -3.4e38f;
    float mx = fmaxf(v0, v1);
#pragma unroll
    for (int o = 16; o > 0; o >>= 1) mx = fmaxf(mx, __shfl_xor(mx, o, 32));
    float e0 = __expf(v0 - mx);
    float e1 = (lane < 16) ? __expf(v1 - mx) : 0.f;
    float s = e0 + e1;
#pragma unroll
    for (int o = 16; o > 0; o >>= 1) s += __shfl_xor(s, o, 32);
    const float inv = 1.f / s;
    float part = e0 * inv * Rs[lane] + ((lane < 16) ? e1 * inv * Rs[lane + 32] : 0.f);
#pragma unroll
    for (int o = 16; o > 0; o >>= 1) part += __shfl_xor(part, o, 32);
    if (lane == 0) out[(size_t)g * OUT_PER_G + 512 + a] = part * 0.125f;
  }
  // top-4 selection (jax.lax.top_k ties -> lowest index via strict >)
  if (tid == 0) {
    bool used[48] = {};
#pragma unroll
    for (int t = 0; t < 4; ++t) {
      float best = -3.4e38f; int bi = 0;
      for (int n = 0; n < 48; ++n)
        if (!used[n] && Sc[n] > best) { best = Sc[n]; bi = n; }
      used[bi] = true; Ti[t] = bi; Tv[t] = best;
    }
  }
  __syncthreads();
  for (int idx = tid; idx < 512; idx += 256) {
    int t = idx >> 7, c = idx & 127;
    out[(size_t)g * OUT_PER_G + idx] = Hs[Ti[t] * HSLD + c] * Tv[t];
  }
}

// ---------------------------------------------------------------------------

static constexpr size_t gat_smem_bytes(int KIN) {
  return (size_t)(128 * (KIN + 2) + 48 * (KIN + 2)) * 2            // Wt, Xh (f16)
       + (size_t)(48 * HSLD + 48 + 48) * 4                          // Hs, Ssrc, Sdst
       + (size_t)(EPG + EPG + 48 + 49 + 48 + EPG) * 4               // edge/CSR ints
       + 16 * 4;                                                    // Red
}
static constexpr size_t final_smem_bytes() {
  return (size_t)(128 * 130 + 48 * 130) * 2
       + (size_t)(48 * HSLD + 48 * HSLD + 48 * ULD + 48 + 48 + 4) * 4
       + 4 * 4;
}

extern "C" void kernel_launch(void* const* d_in, const int* in_sizes, int n_in,
                              void* d_out, int out_size, void* d_ws, size_t ws_size,
                              hipStream_t stream) {
  (void)in_sizes; (void)n_in; (void)out_size; (void)ws_size;
  const float* x     = (const float*)d_in[0];
  const int*   esrc  = (const int*)d_in[1];
  const int*   edst  = (const int*)d_in[2];
  // d_in[3] = batch (implicit in layout, unused)
  const float* W1  = (const float*)d_in[4];
  const float* as1 = (const float*)d_in[5];
  const float* ad1 = (const float*)d_in[6];
  const float* b1  = (const float*)d_in[7];
  const float* W2  = (const float*)d_in[8];
  const float* as2 = (const float*)d_in[9];
  const float* ad2 = (const float*)d_in[10];
  const float* b2  = (const float*)d_in[11];
  const float* W3  = (const float*)d_in[12];
  const float* as3 = (const float*)d_in[13];
  const float* ad3 = (const float*)d_in[14];
  const float* b3  = (const float*)d_in[15];
  const float* l1w = (const float*)d_in[16];
  const float* l1b = (const float*)d_in[17];
  const float* l2w = (const float*)d_in[18];
  const float* l2b = (const float*)d_in[19];
  const float* l3w = (const float*)d_in[20];
  const float* l3b = (const float*)d_in[21];
  const float* pw  = (const float*)d_in[22];
  const float* f1w = (const float*)d_in[23];
  const float* f1b = (const float*)d_in[24];
  const float* f2w = (const float*)d_in[25];
  const float* f2b = (const float*)d_in[26];
  float* out = (float*)d_out;

  // ping-pong node-feature buffers: 2 x 98304*128 f32 = 96 MB
  float* hA = (float*)d_ws;
  float* hB = hA + (size_t)G_GRAPHS * N_NODES * HID;

  dim3 grid(G_GRAPHS), block(256);
  gat_layer<64, true><<<grid, block, gat_smem_bytes(64), stream>>>(
      x, esrc, edst, W1, as1, ad1, b1, l1w, l1b, hA);
  gat_layer<128, true><<<grid, block, gat_smem_bytes(128), stream>>>(
      hA, esrc, edst, W2, as2, ad2, b2, l2w, l2b, hB);
  gat_layer<128, false><<<grid, block, gat_smem_bytes(128), stream>>>(
      hB, esrc, edst, W3, as3, ad3, b3, l3w, l3b, hA);
  final_pool<<<grid, block, final_smem_bytes(), stream>>>(
      hA, pw, f1w, f1b, f2w, f2b, out);
}
